// MambaBlock_s_74646531604855
// MI455X (gfx1250) — matrix operand order repaired and checked by reading, not compiled, not run on hardware
//
#include <hip/hip_runtime.h>
#include <hip/hip_bf16.h>
#include <cmath>
#include <cstdint>

// ---------------------------------------------------------------------------
// Types for CDNA5 WMMA (wave32): 16x16x32 bf16 -> f32 accumulate
// ---------------------------------------------------------------------------
typedef __attribute__((ext_vector_type(16))) __bf16   v16bf;
typedef __attribute__((ext_vector_type(8)))  __bf16   v8bf;
typedef __attribute__((ext_vector_type(8)))  float    v8f;
typedef __attribute__((ext_vector_type(4)))  int      v4i_t;
typedef __attribute__((ext_vector_type(8)))  int      v8i_t;
typedef __attribute__((ext_vector_type(4)))  unsigned v4u_t;

constexpr int D_MODEL = 1024;
constexpr int DIN     = 2048;   // dim_inner
constexpr int NST     = 16;     // d_state
constexpr int DTR     = 64;     // dt_rank
constexpr int SEQ     = 2048;
constexpr int BATCH   = 2;
constexpr int ROWS    = BATCH * SEQ;   // 4096 tokens
constexpr float EPS   = 1e-5f;

// CDNA5 Tensor Data Mover path (TENSOR_LOAD_TO_LDS + TENSORcnt).
#if defined(__AMDGCN__) &&                                                  \
    __has_builtin(__builtin_amdgcn_tensor_load_to_lds) &&                   \
    __has_builtin(__builtin_amdgcn_s_wait_tensorcnt)
#define USE_TDM 1
#else
#define USE_TDM 0
#endif

// CDNA5 async global->LDS path (GLOBAL_LOAD_ASYNC_TO_LDS_B128 + ASYNCcnt).
// Signature (from toolchain diagnostics): (v4i AS1*, v4i AS3*, imm offset, imm cpol).
#if defined(__AMDGCN__) &&                                                  \
    __has_builtin(__builtin_amdgcn_global_load_async_to_lds_b128) &&        \
    __has_builtin(__builtin_amdgcn_s_wait_asynccnt)
#define USE_ASYNC_LDS 1
typedef __attribute__((address_space(1))) v4i_t* gas_ptr;
typedef __attribute__((address_space(3))) v4i_t* las_ptr;
#else
#define USE_ASYNC_LDS 0
#endif

__device__ __forceinline__ float silu_f(float x) {
  return x * (1.0f / (1.0f + __expf(-x)));
}

// RNE f32 -> bf16 (bit-exact, avoids relying on __bf16 cast lowering)
__device__ __forceinline__ __bf16 f2bf(float f) {
  unsigned u = __float_as_uint(f);
  u += 0x7FFFu + ((u >> 16) & 1u);
  union { unsigned short s; __bf16 b; } cv;
  cv.s = (unsigned short)(u >> 16);
  return cv.b;
}

#if USE_TDM
// Issue a TDM 2D tile load: tile_rows x 32 bf16 elements, row stride K elements,
// packed contiguously into LDS at lds_addr (row = 64 bytes).
// D# layout per CDNA5 ISA ch.8: group0 = {count/lds_addr/global_addr/type},
// group1 = {data_size, tensor dims, tile dims, strides}; groups 2-3 zero (2D).
__device__ __forceinline__ void tdm_load_2d(const void* gaddr, unsigned lds_addr,
                                            int tile_rows, int K, int tensor_rows) {
  unsigned long long ga = (unsigned long long)(size_t)gaddr;
  v4u_t g0;
  g0.x = 1u;                                                  // count=1, user mode
  g0.y = lds_addr;                                            // LDS byte address
  g0.z = (unsigned)ga;                                        // global_addr[31:0]
  g0.w = (unsigned)((ga >> 32) & 0x1FFFFFFu) | (2u << 30);    // addr[56:32] | type=2
  unsigned td0 = (unsigned)K;           // tensor_dim0 (elements)
  unsigned td1 = (unsigned)tensor_rows; // tensor_dim1 (rows)
  v8i_t g1;
  g1[0] = 0x10000;                                            // data_size=1 (2 bytes)
  g1[1] = (int)((td0 & 0xFFFFu) << 16);                       // barrier_addr=0 | td0.lo
  g1[2] = (int)((td0 >> 16) | ((td1 & 0xFFFFu) << 16));       // td0.hi | td1.lo
  g1[3] = (int)((td1 >> 16) | (32u << 16));                   // td1.hi | tile_dim0=32
  g1[4] = (int)(tile_rows & 0xFFFF);                          // tile_dim1 | tile_dim2=0
  g1[5] = (int)td0;                                           // tensor_dim0_stride.lo=K
  g1[6] = 0;                                                  // stride0.hi | stride1.lo
  g1[7] = 0;                                                  // stride1.hi
  v4i_t z4 = {0, 0, 0, 0};
#if __clang_major__ >= 23
  v8i_t z8 = {0, 0, 0, 0, 0, 0, 0, 0};
  __builtin_amdgcn_tensor_load_to_lds(g0, g1, z4, z4, z8, 0);
#else
  __builtin_amdgcn_tensor_load_to_lds(g0, g1, z4, z4, 0);
#endif
}
#endif

// ---------------------------------------------------------------------------
// Elementwise fp32 -> bf16 conversion
// ---------------------------------------------------------------------------
__global__ __launch_bounds__(256) void k_convert(const float* __restrict__ src,
                                                 __bf16* __restrict__ dst, int n) {
  int i = blockIdx.x * 256 + threadIdx.x;
  if (i < n) dst[i] = f2bf(src[i]);
}

// convert [rows_src, cols] and zero-pad to [rows_dst, cols]
__global__ __launch_bounds__(256) void k_convert_pad(const float* __restrict__ src,
                                                     __bf16* __restrict__ dst,
                                                     int rows_src, int rows_dst, int cols) {
  int i = blockIdx.x * 256 + threadIdx.x;
  if (i >= rows_dst * cols) return;
  int row = i / cols;
  dst[i] = (row < rows_src) ? f2bf(src[i]) : f2bf(0.0f);
}

// ---------------------------------------------------------------------------
// ada = silu(c) @ ada_w^T + ada_b : [B, 3*D].  One wave32 per output element.
// ---------------------------------------------------------------------------
__global__ __launch_bounds__(256) void k_ada(const float* __restrict__ c,
                                             const float* __restrict__ ada_w,
                                             const float* __restrict__ ada_b,
                                             float* __restrict__ ada) {
  int gw   = (blockIdx.x * 256 + threadIdx.x) >> 5;   // global wave id
  int lane = threadIdx.x & 31;
  if (gw >= BATCH * 3 * D_MODEL) return;
  int b = gw / (3 * D_MODEL);
  int e = gw % (3 * D_MODEL);
  const float* crow = c + (size_t)b * 2 * D_MODEL;
  const float* wrow = ada_w + (size_t)e * 2 * D_MODEL;
  float s = 0.0f;
  for (int k = lane; k < 2 * D_MODEL; k += 32)
    s += silu_f(crow[k]) * wrow[k];
  for (int m = 16; m >= 1; m >>= 1) s += __shfl_xor(s, m, 32);
  if (lane == 0) ada[gw] = s + ada_b[e];
}

// ---------------------------------------------------------------------------
// LayerNorm + AdaLN modulation, emits bf16 activations. Block per token row.
// ---------------------------------------------------------------------------
__global__ __launch_bounds__(256) void k_ln_mod(const float* __restrict__ x,
                                                const float* __restrict__ ln_g,
                                                const float* __restrict__ ln_b,
                                                const float* __restrict__ ada,
                                                __bf16* __restrict__ xnbf) {
  const int tid = threadIdx.x;
  const int r = blockIdx.x;
  const int b = r / SEQ;
  const float* xr = x + (size_t)r * D_MODEL;
  float v[4], s = 0.0f, ss = 0.0f;
#pragma unroll
  for (int i = 0; i < 4; ++i) {
    v[i] = xr[tid + i * 256];
    s += v[i];
    ss += v[i] * v[i];
  }
  __shared__ float sh1[256], sh2[256];
  sh1[tid] = s; sh2[tid] = ss;
  __syncthreads();
  for (int o = 128; o > 0; o >>= 1) {
    if (tid < o) { sh1[tid] += sh1[tid + o]; sh2[tid] += sh2[tid + o]; }
    __syncthreads();
  }
  float mu  = sh1[0] * (1.0f / D_MODEL);
  float var = sh2[0] * (1.0f / D_MODEL) - mu * mu;
  float inv = rsqrtf(var + EPS);
  const float* shift = ada + (size_t)b * 3 * D_MODEL;
  const float* scale = shift + D_MODEL;
#pragma unroll
  for (int i = 0; i < 4; ++i) {
    int col = tid + i * 256;
    float xn = (v[i] - mu) * inv * ln_g[col] + ln_b[col];
    xn = xn * (1.0f + scale[col]) + shift[col];
    xnbf[(size_t)r * D_MODEL + col] = f2bf(xn);
  }
}

// ---------------------------------------------------------------------------
// Tiled bf16 WMMA GEMM: C[M,N] = A[M,K] * B[N,K]^T  (both bf16 row-major,
// K-contiguous => symmetric operand loading), fp32 C.
// Block tile 64(M)x128(N), K-step 32; 8 waves: 4(M) x 2(N); wave = 16x64 (4 wmma).
// Requires M%64==0, N%128==0, K%32==0 (guaranteed by padding).
//
// Fragment layouts used here (contiguous k order, shared by A and B; not the ISA table order):
//  A (16x32 bf16): lane l: m=l&15, half=l>>4; slots 0-15 = k in [half*16, half*16+16).
//  B (32x16 bf16): lane l: n=l&15, half=l>>4; slots 0-15 = k in [half*16, half*16+16).
//  C (16x16 f32):  lane l: n=l&15; vgpr r holds row m = 8*(l>>4) + r.
//
// Data movement, best first:
//  1) TDM descriptor DMA (TENSOR_LOAD_TO_LDS, TENSORcnt): wave 0 issues two
//     2D tile descriptors per K-step; double-buffered, 1 barrier per step.
//  2) GLOBAL_LOAD_ASYNC_TO_LDS_B128 (ASYNCcnt), double-buffered.
//  3) Classic load/ds_store staging, 2 barriers per step.
// ---------------------------------------------------------------------------
__global__ __launch_bounds__(256) void k_gemm_bf16(const __bf16* __restrict__ A,
                                                   const __bf16* __restrict__ B,
                                                   float* __restrict__ C,
                                                   int M, int N, int K) {
  __shared__ alignas(64) __bf16 As[2][64][32];
  __shared__ alignas(64) __bf16 Bs[2][128][32];

  const int tid  = threadIdx.x;
  const int lane = tid & 31;
  const int wave = tid >> 5;
  const int wm   = wave & 3;    // m strip (16 rows)
  const int wn   = wave >> 2;   // n strip (64 cols)
  const int bm   = blockIdx.y * 64;
  const int bn   = blockIdx.x * 128;

  const int ml   = lane & 15;
  const int half = lane >> 4;

  // cooperative tile load assignments (async/fallback paths)
  const int arow = tid >> 2, aseg = tid & 3;   // 64 rows x 4 segs of 8 bf16
  const int brow = tid >> 1, bseg = tid & 1;   // 128 rows x 2 segs of 16 bf16

  const __bf16* gA = &A[(size_t)(bm + arow) * K + aseg * 8];
  const __bf16* gB = &B[(size_t)(bn + brow) * K + bseg * 16];

  v8f acc0 = {}, acc1 = {}, acc2 = {}, acc3 = {};

#if USE_TDM
  const unsigned ldsA0 = (unsigned)(size_t)(las_ptr)(void*)&As[0][0][0];
  const unsigned ldsA1 = (unsigned)(size_t)(las_ptr)(void*)&As[1][0][0];
  const unsigned ldsB0 = (unsigned)(size_t)(las_ptr)(void*)&Bs[0][0][0];
  const unsigned ldsB1 = (unsigned)(size_t)(las_ptr)(void*)&Bs[1][0][0];
  auto issue_tdm = [&](int bu, int k0) {
    if (wave == 0) {
      tdm_load_2d(&A[(size_t)bm * K + k0], bu ? ldsA1 : ldsA0, 64, K, M);
      tdm_load_2d(&B[(size_t)bn * K + k0], bu ? ldsB1 : ldsB0, 128, K, N);
    }
  };
  issue_tdm(0, 0);
  int buf = 0;
  for (int k0 = 0; k0 < K; k0 += 32) {
    __builtin_amdgcn_s_wait_tensorcnt(0);  // wave 0's DMA for `buf` done (others: no-op)
    __syncthreads();                       // publish to all waves
    if (k0 + 32 < K) issue_tdm(buf ^ 1, k0 + 32);

    v16bf af = *(const v16bf*)&As[buf][wm * 16 + ml][half * 16];

    v16bf b0 = *(const v16bf*)&Bs[buf][wn * 64 +  0 + ml][half * 16];
    v16bf b1 = *(const v16bf*)&Bs[buf][wn * 64 + 16 + ml][half * 16];
    v16bf b2 = *(const v16bf*)&Bs[buf][wn * 64 + 32 + ml][half * 16];
    v16bf b3 = *(const v16bf*)&Bs[buf][wn * 64 + 48 + ml][half * 16];

    acc0 = __builtin_amdgcn_wmma_f32_16x16x32_bf16(false, af, false, b0, (short)0, acc0, false, false);
    acc1 = __builtin_amdgcn_wmma_f32_16x16x32_bf16(false, af, false, b1, (short)0, acc1, false, false);
    acc2 = __builtin_amdgcn_wmma_f32_16x16x32_bf16(false, af, false, b2, (short)0, acc2, false, false);
    acc3 = __builtin_amdgcn_wmma_f32_16x16x32_bf16(false, af, false, b3, (short)0, acc3, false, false);

    buf ^= 1;
  }
#elif USE_ASYNC_LDS
  auto issue_tile = [&](int bu, int k0) {
    __builtin_amdgcn_global_load_async_to_lds_b128(
        (gas_ptr)(void*)(gA + k0), (las_ptr)(void*)&As[bu][arow][aseg * 8], 0, 0);
    __builtin_amdgcn_global_load_async_to_lds_b128(
        (gas_ptr)(void*)(gB + k0), (las_ptr)(void*)&Bs[bu][brow][bseg * 16], 0, 0);
    __builtin_amdgcn_global_load_async_to_lds_b128(
        (gas_ptr)(void*)(gB + k0 + 8), (las_ptr)(void*)&Bs[bu][brow][bseg * 16 + 8], 0, 0);
  };
  issue_tile(0, 0);
  int buf = 0;
  for (int k0 = 0; k0 < K; k0 += 32) {
    __builtin_amdgcn_s_wait_asynccnt(0);   // own async-LDS writes for `buf` done
    __syncthreads();                       // everyone's writes for `buf` done
    if (k0 + 32 < K) issue_tile(buf ^ 1, k0 + 32);

    v16bf af = *(const v16bf*)&As[buf][wm * 16 + ml][half * 16];

    v16bf b0 = *(const v16bf*)&Bs[buf][wn * 64 +  0 + ml][half * 16];
    v16bf b1 = *(const v16bf*)&Bs[buf][wn * 64 + 16 + ml][half * 16];
    v16bf b2 = *(const v16bf*)&Bs[buf][wn * 64 + 32 + ml][half * 16];
    v16bf b3 = *(const v16bf*)&Bs[buf][wn * 64 + 48 + ml][half * 16];

    acc0 = __builtin_amdgcn_wmma_f32_16x16x32_bf16(false, af, false, b0, (short)0, acc0, false, false);
    acc1 = __builtin_amdgcn_wmma_f32_16x16x32_bf16(false, af, false, b1, (short)0, acc1, false, false);
    acc2 = __builtin_amdgcn_wmma_f32_16x16x32_bf16(false, af, false, b2, (short)0, acc2, false, false);
    acc3 = __builtin_amdgcn_wmma_f32_16x16x32_bf16(false, af, false, b3, (short)0, acc3, false, false);

    buf ^= 1;
  }
#else
  for (int k0 = 0; k0 < K; k0 += 32) {
    *(v8bf*)&As[0][arow][aseg * 8]   = *(const v8bf*)(gA + k0);
    *(v16bf*)&Bs[0][brow][bseg * 16] = *(const v16bf*)(gB + k0);
    __syncthreads();

    if (k0 + 32 < K) {
      __builtin_prefetch(gA + k0 + 32, 0, 3);
      __builtin_prefetch(gB + k0 + 32, 0, 3);
    }

    v16bf af = *(const v16bf*)&As[0][wm * 16 + ml][half * 16];

    v16bf b0 = *(const v16bf*)&Bs[0][wn * 64 +  0 + ml][half * 16];
    v16bf b1 = *(const v16bf*)&Bs[0][wn * 64 + 16 + ml][half * 16];
    v16bf b2 = *(const v16bf*)&Bs[0][wn * 64 + 32 + ml][half * 16];
    v16bf b3 = *(const v16bf*)&Bs[0][wn * 64 + 48 + ml][half * 16];

    acc0 = __builtin_amdgcn_wmma_f32_16x16x32_bf16(false, af, false, b0, (short)0, acc0, false, false);
    acc1 = __builtin_amdgcn_wmma_f32_16x16x32_bf16(false, af, false, b1, (short)0, acc1, false, false);
    acc2 = __builtin_amdgcn_wmma_f32_16x16x32_bf16(false, af, false, b2, (short)0, acc2, false, false);
    acc3 = __builtin_amdgcn_wmma_f32_16x16x32_bf16(false, af, false, b3, (short)0, acc3, false, false);

    __syncthreads();
  }
#endif

  float* crow = C + (size_t)(bm + wm * 16 + half * 8) * N + bn + wn * 64 + ml;
  v8f accs[4] = {acc0, acc1, acc2, acc3};
#pragma unroll
  for (int i = 0; i < 4; ++i) {
#pragma unroll
    for (int r2 = 0; r2 < 8; ++r2)
      crow[(size_t)r2 * N + i * 16] = accs[i][r2];
  }
}

// ---------------------------------------------------------------------------
// Depthwise causal conv1d (k=4) + bias + silu over x_in half of xz.
// ---------------------------------------------------------------------------
__global__ __launch_bounds__(256) void k_conv(const float* __restrict__ xz,
                                              const float* __restrict__ conv_w,
                                              const float* __restrict__ conv_b,
                                              float* __restrict__ u,
                                              __bf16* __restrict__ ubf) {
  size_t idx = (size_t)blockIdx.x * 256 + threadIdx.x;   // over ROWS*DIN
  int di = (int)(idx % DIN);
  int r  = (int)(idx / DIN);
  int b  = r / SEQ, s = r % SEQ;
  float acc = conv_b[di];
#pragma unroll
  for (int t = 0; t < 4; ++t) {
    int st = s - 3 + t;
    if (st >= 0)
      acc += conv_w[di * 4 + t] * xz[((size_t)(b * SEQ + st)) * (2 * DIN) + di];
  }
  float uv = silu_f(acc);
  u[idx] = uv;
  ubf[idx] = f2bf(uv);
}

// extract dt_r (first 64 cols of padded x_dbl) as bf16 GEMM operand
__global__ __launch_bounds__(256) void k_dtr(const float* __restrict__ xdbl,
                                             __bf16* __restrict__ dtrbf) {
  size_t idx = (size_t)blockIdx.x * 256 + threadIdx.x;   // over ROWS*DTR
  size_t r = idx / DTR, j = idx % DTR;
  dtrbf[idx] = f2bf(xdbl[r * 128 + j]);
}

// dt = softplus(dt_pre + dt_proj_b), in place
__global__ __launch_bounds__(256) void k_softplus(float* __restrict__ dt,
                                                  const float* __restrict__ dt_proj_b) {
  size_t idx = (size_t)blockIdx.x * 256 + threadIdx.x;   // over ROWS*DIN
  int di = (int)(idx % DIN);
  float v = dt[idx] + dt_proj_b[di];
  dt[idx] = (v > 20.0f) ? v : log1pf(__expf(v));
}

// ---------------------------------------------------------------------------
// Selective scan: one thread per (batch, channel); h[16] in registers;
// B_t / C_t broadcast through LDS. Emits gated bf16 output for out_proj.
// ---------------------------------------------------------------------------
__global__ __launch_bounds__(256) void k_scan(const float* __restrict__ dt,
                                              const float* __restrict__ u,
                                              const float* __restrict__ xdbl,
                                              const float* __restrict__ xz,
                                              const float* __restrict__ A_log,
                                              const float* __restrict__ D_param,
                                              __bf16* __restrict__ ybf) {
  const int b  = blockIdx.x >> 3;                 // DIN/256 = 8 blocks per batch
  const int di = (blockIdx.x & 7) * 256 + threadIdx.x;
  float a[NST], h[NST];
#pragma unroll
  for (int n = 0; n < NST; ++n) {
    a[n] = -__expf(A_log[(size_t)di * NST + n]);
    h[n] = 0.0f;
  }
  const float Dv = D_param[di];
  __shared__ float sB[NST], sC[NST];
  for (int s = 0; s < SEQ; ++s) {
    size_t r = (size_t)b * SEQ + s;
    if (threadIdx.x < NST)
      sB[threadIdx.x] = xdbl[r * 128 + DTR + threadIdx.x];
    else if (threadIdx.x < 2 * NST)
      sC[threadIdx.x - NST] = xdbl[r * 128 + DTR + NST + (threadIdx.x - NST)];
    __syncthreads();
    float dtv = dt[r * DIN + di];
    float uv  = u[r * DIN + di];
    float zv  = xz[r * (2 * DIN) + DIN + di];
    float du  = dtv * uv;
    float y = 0.0f;
#pragma unroll
    for (int n = 0; n < NST; ++n) {
      h[n] = h[n] * __expf(dtv * a[n]) + du * sB[n];
      y += h[n] * sC[n];
    }
    y = (y + uv * Dv) * silu_f(zv);
    ybf[r * DIN + di] = f2bf(y);
    __syncthreads();
  }
}

// out = x + gate * y_ssm
__global__ __launch_bounds__(256) void k_final(const float* __restrict__ x,
                                               const float* __restrict__ ada,
                                               const float* __restrict__ yout,
                                               float* __restrict__ out) {
  size_t idx = (size_t)blockIdx.x * 256 + threadIdx.x;   // over ROWS*D_MODEL
  int e = (int)(idx % D_MODEL);
  int r = (int)(idx / D_MODEL);
  int b = r / SEQ;
  float gate = ada[(size_t)b * 3 * D_MODEL + 2 * D_MODEL + e];
  out[idx] = x[idx] + gate * yout[idx];
}

// ---------------------------------------------------------------------------
extern "C" void kernel_launch(void* const* d_in, const int* in_sizes, int n_in,
                              void* d_out, int out_size, void* d_ws, size_t ws_size,
                              hipStream_t stream) {
  (void)in_sizes; (void)n_in; (void)out_size; (void)ws_size;
  const float* x         = (const float*)d_in[0];
  const float* c         = (const float*)d_in[1];
  const float* ln_g      = (const float*)d_in[2];
  const float* ln_b      = (const float*)d_in[3];
  const float* ada_w     = (const float*)d_in[4];
  const float* ada_b     = (const float*)d_in[5];
  const float* in_proj_w = (const float*)d_in[6];
  const float* conv_w    = (const float*)d_in[7];
  const float* conv_b    = (const float*)d_in[8];
  const float* x_proj_w  = (const float*)d_in[9];
  const float* dt_proj_w = (const float*)d_in[10];
  const float* dt_proj_b = (const float*)d_in[11];
  const float* A_log     = (const float*)d_in[12];
  const float* D_param   = (const float*)d_in[13];
  const float* out_proj_w= (const float*)d_in[14];
  float* out = (float*)d_out;

  char* p = (char*)d_ws;
  auto alloc = [&](size_t bytes) -> char* {
    char* r = p; p += (bytes + 255) & ~(size_t)255; return r;
  };
  float*  ada    = (float*) alloc((size_t)BATCH * 3 * D_MODEL * 4);
  __bf16* xnbf   = (__bf16*)alloc((size_t)ROWS * D_MODEL * 2);
  __bf16* w_in   = (__bf16*)alloc((size_t)2 * DIN * D_MODEL * 2);
  float*  xz     = (float*) alloc((size_t)ROWS * 2 * DIN * 4);
  float*  ubuf   = (float*) alloc((size_t)ROWS * DIN * 4);
  __bf16* ubf    = (__bf16*)alloc((size_t)ROWS * DIN * 2);
  __bf16* w_xp   = (__bf16*)alloc((size_t)128 * DIN * 2);      // x_proj padded 96->128
  float*  xdbl   = (float*) alloc((size_t)ROWS * 128 * 4);
  __bf16* dtrbf  = (__bf16*)alloc((size_t)ROWS * DTR * 2);
  __bf16* w_dt   = (__bf16*)alloc((size_t)DIN * DTR * 2);
  float*  dtbuf  = (float*) alloc((size_t)ROWS * DIN * 4);
  __bf16* ybf    = (__bf16*)alloc((size_t)ROWS * DIN * 2);
  __bf16* w_out  = (__bf16*)alloc((size_t)D_MODEL * DIN * 2);
  float*  yout   = (float*) alloc((size_t)ROWS * D_MODEL * 4);

  // weight precision conversion (deterministic per call)
  k_convert<<<(2 * DIN * D_MODEL) / 256, 256, 0, stream>>>(in_proj_w, w_in, 2 * DIN * D_MODEL);
  k_convert<<<(DIN * DTR) / 256, 256, 0, stream>>>(dt_proj_w, w_dt, DIN * DTR);
  k_convert<<<(D_MODEL * DIN) / 256, 256, 0, stream>>>(out_proj_w, w_out, D_MODEL * DIN);
  k_convert_pad<<<(128 * DIN) / 256, 256, 0, stream>>>(x_proj_w, w_xp, DTR + 2 * NST, 128, DIN);

  // conditioning + layernorm
  k_ada<<<(BATCH * 3 * D_MODEL * 32) / 256, 256, 0, stream>>>(c, ada_w, ada_b, ada);
  k_ln_mod<<<ROWS, 256, 0, stream>>>(x, ln_g, ln_b, ada, xnbf);

  // in_proj: [4096,1024] x [4096,1024]^T -> [4096,4096]
  k_gemm_bf16<<<dim3((2 * DIN) / 128, ROWS / 64), 256, 0, stream>>>(
      xnbf, w_in, xz, ROWS, 2 * DIN, D_MODEL);

  k_conv<<<(ROWS * DIN) / 256, 256, 0, stream>>>(xz, conv_w, conv_b, ubuf, ubf);

  // x_proj: [4096,2048] x [128,2048]^T -> [4096,128]
  k_gemm_bf16<<<dim3(1, ROWS / 64), 256, 0, stream>>>(ubf, w_xp, xdbl, ROWS, 128, DIN);

  k_dtr<<<(ROWS * DTR) / 256, 256, 0, stream>>>(xdbl, dtrbf);

  // dt_proj: [4096,64] x [2048,64]^T -> [4096,2048]
  k_gemm_bf16<<<dim3(DIN / 128, ROWS / 64), 256, 0, stream>>>(
      dtrbf, w_dt, dtbuf, ROWS, DIN, DTR);

  k_softplus<<<(ROWS * DIN) / 256, 256, 0, stream>>>(dtbuf, dt_proj_b);

  k_scan<<<BATCH * (DIN / 256), 256, 0, stream>>>(dtbuf, ubuf, xdbl, xz, A_log, D_param, ybf);

  // out_proj: [4096,2048] x [1024,2048]^T -> [4096,1024]
  k_gemm_bf16<<<dim3(D_MODEL / 128, ROWS / 64), 256, 0, stream>>>(
      ybf, w_out, yout, ROWS, D_MODEL, DIN);

  k_final<<<(ROWS * D_MODEL) / 256, 256, 0, stream>>>(x, ada, yout, out);
}
